// Attention_1717986918775
// MI455X (gfx1250) — matrix operand order repaired and checked by reading, not compiled, not run on hardware
//
#include <hip/hip_runtime.h>
#include <math.h>

// ---------------- types ----------------
typedef __bf16 bf16;
typedef __bf16  v16bf __attribute__((ext_vector_type(16)));
typedef float   v8f   __attribute__((ext_vector_type(8)));
typedef float   f32x4 __attribute__((ext_vector_type(4)));
typedef float   f32x2 __attribute__((ext_vector_type(2)));

union Frag16 { v16bf v; f32x4 f4[2]; };

struct alignas(8) bf16x4 { bf16 a, b, c, d; };
struct alignas(4) bf16x2 { bf16 x, y; };

// ---------------- constants ----------------
#define BB 4
#define TT 2048
#define EE 1024
#define HH 16
#define DD 64

// ---------------- CDNA5 async copy helpers ----------------
__device__ __forceinline__ uint32_t lds_offset(const void* p) {
  // generic pointers into the LDS aperture carry the LDS byte offset in [31:0]
  return (uint32_t)(uintptr_t)p;
}
__device__ __forceinline__ void async_load_b128(uint32_t lds_off, const void* g) {
  asm volatile("global_load_async_to_lds_b128 %0, %1, off"
               :: "v"(lds_off), "v"((uint64_t)(uintptr_t)g)
               : "memory");
}
__device__ __forceinline__ void wait_async_le4() {
  asm volatile("s_wait_asynccnt 4" ::: "memory");
}
__device__ __forceinline__ void wait_async_le2() {
  asm volatile("s_wait_asynccnt 2" ::: "memory");
}
__device__ __forceinline__ void wait_async_0() {
  asm volatile("s_wait_asynccnt 0" ::: "memory");
}

// ---------------- WMMA helpers ----------------
__device__ __forceinline__ v8f wmma_bf16(v16bf a, v16bf b, v8f c) {
  return __builtin_amdgcn_wmma_f32_16x16x32_bf16(
      false, a, false, b, (short)0, c, false, false);
}

// A operand: 16x32 bf16, row m = lane&15, element e:
//   e<8 -> k = 8*half + e ; e>=8 -> k = 16 + 8*half + (e-8)
__device__ __forceinline__ v16bf load_fragA(const bf16* base, int ld, int row0,
                                            int k0, int lane) {
  int m = lane & 15, half = lane >> 4;
  const bf16* p = base + (size_t)(row0 + m) * ld + k0 + 8 * half;
  Frag16 f;
  f.f4[0] = *(const f32x4*)(p);
  f.f4[1] = *(const f32x4*)(p + 16);
  return f.v;
}

// B operand: 32x16 bf16, col n = lane&15, element e:
//   e<8 -> k = 8*half + e ; e>=8 -> k = 16 + 8*half + (e-8)
__device__ __forceinline__ v16bf load_fragB(const bf16* base, int ld, int n0,
                                            int k0, int lane) {
  int n = lane & 15, half = lane >> 4;
  const bf16* p = base + (size_t)(n0 + n) * ld + k0 + 8 * half;
  Frag16 f;
  f.f4[0] = *(const f32x4*)(p);
  f.f4[1] = *(const f32x4*)(p + 16);
  return f.v;
}

// ---------------- elementwise converts ----------------
__global__ __launch_bounds__(256) void convert_f32_bf16_kernel(
    const float* __restrict__ X, bf16* __restrict__ Xb, int n4) {
  int i = blockIdx.x * 256 + threadIdx.x;
  if (i >= n4) return;
  f32x4 v = *(const f32x4*)(X + (size_t)i * 4);
  bf16x4 o = {(bf16)v[0], (bf16)v[1], (bf16)v[2], (bf16)v[3]};
  *(bf16x4*)(Xb + (size_t)i * 4) = o;
}

// W [K][N] f32 -> Wt [N][K] bf16
__global__ __launch_bounds__(256) void transpose_f32_bf16_kernel(
    const float* __restrict__ W, bf16* __restrict__ Wt, int K, int N) {
  size_t idx = (size_t)blockIdx.x * 256 + threadIdx.x;
  if (idx >= (size_t)K * N) return;
  int k = (int)(idx / N), n = (int)(idx % N);
  Wt[(size_t)n * K + k] = (bf16)W[idx];
}

// ---------------- bf16 GEMM: C[M][N] = A[M][K] @ Bt[N][K]^T + bias ----------------
// Block = 8 waves arranged 4(M) x 2(N): 128x128 macro-tile, 32x64 per wave.
// Async double-buffered 32-wide k stages: A 128x32, B 128x32 (rows padded to
// 40 el = 80B for bank-conflict-free b128 reads). 4 async loads/thread/stage.
#define G_STRIDE 40
__device__ __forceinline__ void gemm_issue_stage(
    const bf16* __restrict__ A, const bf16* __restrict__ Bt, int K,
    int mBase, int nBase, int kStage, bf16* tA, bf16* tB, int tid) {
#pragma unroll
  for (int i = 0; i < 2; ++i) {            // A tile: 128 rows x 32 el = 512 x 16B
    int c = tid + i * 256;
    int row = c >> 2, off8 = (c & 3) * 8;
    async_load_b128(lds_offset(tA + row * G_STRIDE + off8),
                    A + (size_t)(mBase + row) * K + kStage + off8);
  }
#pragma unroll
  for (int i = 0; i < 2; ++i) {            // B tile: 128 rows x 32 el = 512 x 16B
    int c = tid + i * 256;
    int row = c >> 2, off8 = (c & 3) * 8;
    async_load_b128(lds_offset(tB + row * G_STRIDE + off8),
                    Bt + (size_t)(nBase + row) * K + kStage + off8);
  }
}

__global__ __launch_bounds__(256) void gemm_bf16_kernel(
    const bf16* __restrict__ A, const bf16* __restrict__ Bt,
    const float* __restrict__ bias, float* __restrict__ C, int M, int N, int K) {
  __shared__ bf16 tA[2][128 * G_STRIDE];
  __shared__ bf16 tB[2][128 * G_STRIDE];
  int tid = threadIdx.x;
  int lane = tid & 31, wslot = tid >> 5;
  int wm = wslot >> 1, wn = wslot & 1;     // 4 waves in M, 2 in N
  int NQ = N >> 7;                         // 128-wide column blocks
  int mBase = (blockIdx.x / NQ) * 128;
  int nBase = (blockIdx.x % NQ) * 128;

  gemm_issue_stage(A, Bt, K, mBase, nBase, 0, tA[0], tB[0], tid);

  v8f acc00{}, acc01{}, acc02{}, acc03{};
  v8f acc10{}, acc11{}, acc12{}, acc13{};
  int NS = K >> 5;                         // 32-wide k stages
#pragma unroll 1
  for (int s = 0; s < NS; ++s) {
    if (s + 1 < NS) {
      gemm_issue_stage(A, Bt, K, mBase, nBase, (s + 1) * 32,
                       tA[(s + 1) & 1], tB[(s + 1) & 1], tid);
      wait_async_le4();                    // stage s complete (in-order)
    } else {
      wait_async_0();
    }
    __syncthreads();
    const bf16* bufA = tA[s & 1];
    const bf16* bufB = tB[s & 1];
    // 6 fragment loads feed 8 WMMAs (full A/B cross reuse)
    v16bf a0 = load_fragA(bufA, G_STRIDE, wm * 32, 0, lane);
    v16bf a1 = load_fragA(bufA, G_STRIDE, wm * 32 + 16, 0, lane);
    v16bf b0 = load_fragB(bufB, G_STRIDE, wn * 64 +  0, 0, lane);
    v16bf b1 = load_fragB(bufB, G_STRIDE, wn * 64 + 16, 0, lane);
    v16bf b2 = load_fragB(bufB, G_STRIDE, wn * 64 + 32, 0, lane);
    v16bf b3 = load_fragB(bufB, G_STRIDE, wn * 64 + 48, 0, lane);
    acc00 = wmma_bf16(a0, b0, acc00);
    acc10 = wmma_bf16(a1, b0, acc10);
    acc01 = wmma_bf16(a0, b1, acc01);
    acc11 = wmma_bf16(a1, b1, acc11);
    acc02 = wmma_bf16(a0, b2, acc02);
    acc12 = wmma_bf16(a1, b2, acc12);
    acc03 = wmma_bf16(a0, b3, acc03);
    acc13 = wmma_bf16(a1, b3, acc13);
    __syncthreads();                       // buffer safe to overwrite
  }

  int half = lane >> 4, n0 = lane & 15;
  int colBase = nBase + wn * 64 + n0;
  float bs0 = bias[colBase], bs1 = bias[colBase + 16];
  float bs2 = bias[colBase + 32], bs3 = bias[colBase + 48];
#pragma unroll
  for (int r = 0; r < 8; ++r) {
    int row0 = mBase + wm * 32 + r + 8 * half;  // subtile 0
    size_t o0 = (size_t)row0 * N + colBase;
    C[o0 +  0] = acc00[r] + bs0;
    C[o0 + 16] = acc01[r] + bs1;
    C[o0 + 32] = acc02[r] + bs2;
    C[o0 + 48] = acc03[r] + bs3;
    size_t o1 = (size_t)(row0 + 16) * N + colBase;  // subtile 1
    C[o1 +  0] = acc10[r] + bs0;
    C[o1 + 16] = acc11[r] + bs1;
    C[o1 + 32] = acc12[r] + bs2;
    C[o1 + 48] = acc13[r] + bs3;
  }
}

// ---------------- per-head LayerNorm + layout ----------------
__device__ __forceinline__ float waveAllSum(float v) {
  v += __shfl_xor(v, 1, 32);
  v += __shfl_xor(v, 2, 32);
  v += __shfl_xor(v, 4, 32);
  v += __shfl_xor(v, 8, 32);
  v += __shfl_xor(v, 16, 32);
  return v;
}

__global__ __launch_bounds__(256) void ln_qk_kernel(
    const float* __restrict__ qkv,
    const float* __restrict__ qg, const float* __restrict__ qb,
    const float* __restrict__ kg, const float* __restrict__ kb,
    bf16* __restrict__ Qh, bf16* __restrict__ Kh, bf16* __restrict__ Vt) {
  int lane = threadIdx.x & 31;
  int wave = blockIdx.x * (blockDim.x >> 5) + (threadIdx.x >> 5);
  if (wave >= BB * TT * HH) return;
  int h = wave % HH;
  int bt = wave / HH;
  int b = bt / TT, t = bt % TT;
  int d0 = lane * 2;

  size_t rowOff = (size_t)bt * (3 * EE) + h * DD;
  size_t hOff = ((size_t)(b * HH + h) * TT + t) * DD;

  {  // q layernorm
    f32x2 x2 = *(const f32x2*)(qkv + rowOff + d0);
    float mean = waveAllSum(x2[0] + x2[1]) * (1.0f / DD);
    float dx0 = x2[0] - mean, dx1 = x2[1] - mean;
    float var = waveAllSum(dx0 * dx0 + dx1 * dx1) * (1.0f / DD);
    float rstd = rsqrtf(var + 1e-5f);
    bf16x2 o = {(bf16)(dx0 * rstd * qg[d0] + qb[d0]),
                (bf16)(dx1 * rstd * qg[d0 + 1] + qb[d0 + 1])};
    *(bf16x2*)(Qh + hOff + d0) = o;
  }
  {  // k layernorm
    f32x2 x2 = *(const f32x2*)(qkv + rowOff + EE + d0);
    float mean = waveAllSum(x2[0] + x2[1]) * (1.0f / DD);
    float dx0 = x2[0] - mean, dx1 = x2[1] - mean;
    float var = waveAllSum(dx0 * dx0 + dx1 * dx1) * (1.0f / DD);
    float rstd = rsqrtf(var + 1e-5f);
    bf16x2 o = {(bf16)(dx0 * rstd * kg[d0] + kb[d0]),
                (bf16)(dx1 * rstd * kg[d0 + 1] + kb[d0 + 1])};
    *(bf16x2*)(Kh + hOff + d0) = o;
  }
  {  // v -> [B,H,D,T]
    f32x2 x2 = *(const f32x2*)(qkv + rowOff + 2 * EE + d0);
    size_t vb = (size_t)(b * HH + h) * DD * TT + t;
    Vt[vb + (size_t)(d0)     * TT] = (bf16)x2[0];
    Vt[vb + (size_t)(d0 + 1) * TT] = (bf16)x2[1];
  }
}

// ---------------- flash attention ----------------
// Block = 8 waves = 8 adjacent 16-row q tiles of one (b,h).
// K/V tiles async-staged into LDS, double-buffered, shared by all 8 waves.
#define P_STRIDE 40   // 80B rows: 16B aligned, conflict-free
#define K_STRIDE 72   // 144B rows

__device__ __forceinline__ void attn_issue_stage(
    const bf16* __restrict__ Kb, const bf16* __restrict__ Vb, int s0,
    bf16* tK, bf16* tV, int tid) {
  {  // K tile: 32 rows (s) x 64 el (d) = 256 x 16B
    int row = tid >> 3, off8 = (tid & 7) * 8;
    async_load_b128(lds_offset(tK + row * K_STRIDE + off8),
                    Kb + (size_t)(s0 + row) * DD + off8);
  }
  {  // V tile: 64 rows (d) x 32 el (s) = 256 x 16B
    int row = tid >> 2, off8 = (tid & 3) * 8;
    async_load_b128(lds_offset(tV + row * P_STRIDE + off8),
                    Vb + (size_t)row * TT + s0 + off8);
  }
}

__global__ __launch_bounds__(256) void attn_kernel(
    const bf16* __restrict__ Qh, const bf16* __restrict__ Kh,
    const bf16* __restrict__ Vt, bf16* __restrict__ Y) {
  __shared__ bf16 tK[2][32 * K_STRIDE];
  __shared__ bf16 tV[2][64 * P_STRIDE];
  __shared__ bf16 tP[8][16 * P_STRIDE];
  int tid = threadIdx.x;
  int lane = tid & 31, wslot = tid >> 5;
  int bh = blockIdx.x >> 4;                 // 16 blocks per (b,h)
  int qt = (blockIdx.x & 15) * 8 + wslot;
  int b = bh >> 4, h = bh & 15;
  int half = lane >> 4, ln16 = lane & 15;

  const bf16* Qb = Qh + (size_t)bh * TT * DD;
  const bf16* Kb = Kh + (size_t)bh * TT * DD;
  const bf16* Vb = Vt + (size_t)bh * DD * TT;
  bf16* P = &tP[wslot][0];

  v16bf qa0 = load_fragA(Qb, DD, qt * 16, 0, lane);
  v16bf qa1 = load_fragA(Qb, DD, qt * 16, 32, lane);

  v8f o0{}, o1{}, o2{}, o3{};
  float rowMax[8], rowSum[8];
#pragma unroll
  for (int r = 0; r < 8; ++r) { rowMax[r] = -INFINITY; rowSum[r] = 0.0f; }

  const float scale = 0.125f;               // 1/sqrt(64)
  const int NS = TT / 32;                   // 64 stages

  attn_issue_stage(Kb, Vb, 0, tK[0], tV[0], tid);

#pragma unroll 1
  for (int s = 0; s < NS; ++s) {
    if (s + 1 < NS) {
      attn_issue_stage(Kb, Vb, (s + 1) * 32, tK[(s + 1) & 1], tV[(s + 1) & 1], tid);
      wait_async_le2();                     // stage s landed (in-order)
    } else {
      wait_async_0();
    }
    __syncthreads();
    const bf16* bK = tK[s & 1];
    const bf16* bV = tV[s & 1];

    // S tiles: 16 q-rows x 32 s-cols, K = d = 64
    v8f S0{}, S1{};
    {
      v16bf k00 = load_fragB(bK, K_STRIDE, 0, 0, lane);
      v16bf k01 = load_fragB(bK, K_STRIDE, 0, 32, lane);
      v16bf k10 = load_fragB(bK, K_STRIDE, 16, 0, lane);
      v16bf k11 = load_fragB(bK, K_STRIDE, 16, 32, lane);
      S0 = wmma_bf16(qa0, k00, S0);
      S0 = wmma_bf16(qa1, k01, S0);
      S1 = wmma_bf16(qa0, k10, S1);
      S1 = wmma_bf16(qa1, k11, S1);
    }

    // online softmax (row m = r + 8*half lives in one 16-lane group)
#pragma unroll
    for (int r = 0; r < 8; ++r) {
      float a = S0[r] * scale, c = S1[r] * scale;
      float mx = fmaxf(a, c);
      mx = fmaxf(mx, __shfl_xor(mx, 1, 32));
      mx = fmaxf(mx, __shfl_xor(mx, 2, 32));
      mx = fmaxf(mx, __shfl_xor(mx, 4, 32));
      mx = fmaxf(mx, __shfl_xor(mx, 8, 32));
      float mnew = fmaxf(rowMax[r], mx);
      float alpha = __expf(rowMax[r] - mnew);
      float p0 = __expf(a - mnew);
      float p1 = __expf(c - mnew);
      float ps = p0 + p1;
      ps += __shfl_xor(ps, 1, 32);
      ps += __shfl_xor(ps, 2, 32);
      ps += __shfl_xor(ps, 4, 32);
      ps += __shfl_xor(ps, 8, 32);
      rowSum[r] = rowSum[r] * alpha + ps;
      rowMax[r] = mnew;
      o0[r] *= alpha; o1[r] *= alpha; o2[r] *= alpha; o3[r] *= alpha;
      int row = r + 8 * half;
      P[row * P_STRIDE + ln16] = (bf16)p0;
      P[row * P_STRIDE + 16 + ln16] = (bf16)p1;
    }

    // O += P @ V (per-wave private P slice; DS ops in-order per wave)
    v16bf pa = load_fragA(P, P_STRIDE, 0, 0, lane);
    v16bf vb0 = load_fragB(bV, P_STRIDE,  0, 0, lane);
    v16bf vb1 = load_fragB(bV, P_STRIDE, 16, 0, lane);
    v16bf vb2 = load_fragB(bV, P_STRIDE, 32, 0, lane);
    v16bf vb3 = load_fragB(bV, P_STRIDE, 48, 0, lane);
    o0 = wmma_bf16(pa, vb0, o0);
    o1 = wmma_bf16(pa, vb1, o1);
    o2 = wmma_bf16(pa, vb2, o2);
    o3 = wmma_bf16(pa, vb3, o3);
    __syncthreads();                        // K/V buffer safe to overwrite
  }

  // normalize, write Y bf16 in [B,T,E] layout
#pragma unroll
  for (int r = 0; r < 8; ++r) {
    float inv = 1.0f / rowSum[r];
    int tRow = qt * 16 + r + 8 * half;
    size_t base = ((size_t)(b * TT + tRow)) * EE + h * DD + ln16;
    Y[base +  0] = (bf16)(o0[r] * inv);
    Y[base + 16] = (bf16)(o1[r] * inv);
    Y[base + 32] = (bf16)(o2[r] * inv);
    Y[base + 48] = (bf16)(o3[r] * inv);
  }
}

// ---------------- launch ----------------
extern "C" void kernel_launch(void* const* d_in, const int* in_sizes, int n_in,
                              void* d_out, int out_size, void* d_ws, size_t ws_size,
                              hipStream_t stream) {
  const float* x     = (const float*)d_in[0];
  const float* Wqkv  = (const float*)d_in[1];
  const float* bqkv  = (const float*)d_in[2];
  const float* qg    = (const float*)d_in[3];
  const float* qb    = (const float*)d_in[4];
  const float* kg    = (const float*)d_in[5];
  const float* kbeta = (const float*)d_in[6];
  const float* Wproj = (const float*)d_in[7];
  const float* bproj = (const float*)d_in[8];
  float* out = (float*)d_out;

  char* ws = (char*)d_ws;
  const size_t MB = 1ull << 20;
  float* qkv   = (float*)(ws + 0);          // 96 MB f32 [B,T,3E] (dead after LN)
  bf16* Yb     = (bf16*)(ws + 0);           // 16 MB bf16, aliases dead qkv
  bf16* Xb     = (bf16*)(ws + 96 * MB);     // 16 MB
  bf16* WqkvT  = (bf16*)(ws + 112 * MB);    // 6 MB  [3E][E]
  bf16* WprojT = (bf16*)(ws + 118 * MB);    // 2 MB  [E][E]
  bf16* Qh     = (bf16*)(ws + 120 * MB);    // 16 MB [B,H,T,D]
  bf16* Kh     = (bf16*)(ws + 136 * MB);    // 16 MB [B,H,T,D]
  bf16* Vt     = (bf16*)(ws + 152 * MB);    // 16 MB [B,H,D,T]

  const int M = BB * TT;  // 8192

  convert_f32_bf16_kernel<<<(M * EE / 4 + 255) / 256, 256, 0, stream>>>(x, Xb, M * EE / 4);
  transpose_f32_bf16_kernel<<<(EE * 3 * EE + 255) / 256, 256, 0, stream>>>(Wqkv, WqkvT, EE, 3 * EE);
  transpose_f32_bf16_kernel<<<(EE * EE + 255) / 256, 256, 0, stream>>>(Wproj, WprojT, EE, EE);

  // QKV GEMM: 128x128 macro-tiles
  gemm_bf16_kernel<<<(M / 128) * (3 * EE / 128), 256, 0, stream>>>(
      Xb, WqkvT, bqkv, qkv, M, 3 * EE, EE);

  // per-head LN(q,k) + head layouts
  ln_qk_kernel<<<(BB * TT * HH) / 8, 256, 0, stream>>>(qkv, qg, qb, kg, kbeta, Qh, Kh, Vt);

  // flash attention: 1024 blocks x 8 waves
  attn_kernel<<<BB * HH * (TT / 128), 256, 0, stream>>>(Qh, Kh, Vt, Yb);

  // output projection
  gemm_bf16_kernel<<<(M / 128) * (EE / 128), 256, 0, stream>>>(
      Yb, WprojT, bproj, out, M, EE, EE);
}
